// THLSTMCell_57784490000883
// MI455X (gfx1250) — matrix operand order repaired and checked by reading, not compiled, not run on hardware
//
#include <hip/hip_runtime.h>
#include <hip/hip_bf16.h>

typedef __attribute__((ext_vector_type(16))) __bf16 v16bf;
typedef __attribute__((ext_vector_type(8)))  __bf16 v8bf;
typedef __attribute__((ext_vector_type(8)))  float  v8f;
typedef __attribute__((ext_vector_type(4)))  float  v4f;
typedef __attribute__((ext_vector_type(4)))  unsigned short v4u16;
typedef unsigned short u16;

#define BATCH 65536
#define IN    128
#define HID   256
#define BM    128     // batch rows per workgroup
#define MT    2       // M tiles per wave (2*16 = 32 rows)
#define NT    4       // N tiles per wave (4*16 = 64 cols)
#define THREADS 512   // 16 waves = 4 mPairs x 4 nGroups

// ---------- helpers ----------
static __device__ __forceinline__ u16 f2bf_bits(float f) {
  unsigned u = __builtin_bit_cast(unsigned, f);
  unsigned r = u + 0x7FFFu + ((u >> 16) & 1u);   // round-to-nearest-even
  return (u16)(r >> 16);
}
static __device__ __forceinline__ float bf2f(u16 h) {
  unsigned u = ((unsigned)h) << 16;
  return __builtin_bit_cast(float, u);
}
static __device__ __forceinline__ float sigm(float x) {
  return 1.0f / (1.0f + __expf(-x));
}

// ---------- weight f32 -> bf16 conversion ----------
__global__ void cvt_f32_bf16(const float* __restrict__ src,
                             u16* __restrict__ dst, int n) {
  int i = blockIdx.x * blockDim.x + threadIdx.x;
  int stride = gridDim.x * blockDim.x;
  for (; i < n; i += stride) dst[i] = f2bf_bits(src[i]);
}

// ---------- dual-M tile GEMM ----------
// acc[mt][nt] += A(16xK, rows (mA+mt)*16.., from LDS) * W^T(Kx16 from global)
// A: LDS bf16 [BM][LDK] row-major. W: global bf16 [HID][LDK] row-major.
// Each k-step: 2 A-fragments + 4 B-fragments feed 8 WMMAs (0.75KB/WMMA).
template <int LDK>
static __device__ __forceinline__ void gemm_tile(
    v8f acc[MT][NT], const u16* __restrict__ shA, const u16* __restrict__ W,
    int lane, int mA, int nBase) {
  const int aRow = mA * 16 + (lane & 15);
  const int aSel = (lane >> 4) * 16;   // A per-lane K offset (contiguous, matches B)
  const int bSel = (lane >> 4) * 16;   // B per-lane K offset (contiguous, matches A)
  const int nCol = lane & 15;
  const u16* aPtr = shA + aRow * LDK + aSel;
  const u16* bPtr = W + (size_t)(nBase + nCol) * LDK + bSel;
#pragma unroll
  for (int k0 = 0; k0 < LDK; k0 += 32) {
    v16bf a[MT];
#pragma unroll
    for (int mt = 0; mt < MT; ++mt) {
      const u16* ap = aPtr + (size_t)mt * 16 * LDK + k0;
      v8bf a0 = *(const v8bf*)(ap);        // K = k0 + sel + [0..7]
      v8bf a1 = *(const v8bf*)(ap + 8);    // K = k0 + sel + [8..15]
      a[mt] = __builtin_shufflevector(a0, a1, 0, 1, 2, 3, 4, 5, 6, 7,
                                      8, 9, 10, 11, 12, 13, 14, 15);
    }
#pragma unroll
    for (int nt = 0; nt < NT; ++nt) {
      v16bf b = *(const v16bf*)(bPtr + (size_t)nt * 16 * LDK + k0);
#pragma unroll
      for (int mt = 0; mt < MT; ++mt)
        acc[mt][nt] = __builtin_amdgcn_wmma_f32_16x16x32_bf16(
            false, a[mt], false, b, (short)0, acc[mt][nt], false, false);
    }
  }
}

static __device__ __forceinline__ void zero_acc(v8f acc[MT][NT]) {
#pragma unroll
  for (int mt = 0; mt < MT; ++mt)
#pragma unroll
    for (int nt = 0; nt < NT; ++nt) acc[mt][nt] = (v8f)0.0f;
}

// all three K segments of one gate: h (K=256), x (K=128), s (K=256)
static __device__ __forceinline__ void gate_gemm(
    v8f acc[MT][NT], int g, const u16* sh_h, const u16* sh_x, const u16* sh_s,
    const u16* __restrict__ Whb, const u16* __restrict__ Wxb,
    const u16* __restrict__ Wsb, int lane, int mA, int nBase) {
  gemm_tile<HID>(acc, sh_h, Whb + (size_t)g * HID * HID, lane, mA, nBase);
  gemm_tile<IN >(acc, sh_x, Wxb + (size_t)g * HID * IN,  lane, mA, nBase);
  gemm_tile<HID>(acc, sh_s, Wsb + (size_t)(g - 1) * HID * HID, lane, mA, nBase);
}

// ---------- fused THLSTM cell ----------
__global__ __launch_bounds__(THREADS)
void thlstm_fused(const float* __restrict__ x_t, const float* __restrict__ delta_t,
                  const float* __restrict__ h_prev, const float* __restrict__ c_prev,
                  const float* __restrict__ Wst, const float* __restrict__ bias,
                  const u16* __restrict__ Whb,  // [6][HID][HID] bf16
                  const u16* __restrict__ Wxb,  // [6][HID][IN ] bf16
                  const u16* __restrict__ Wsb,  // [5][HID][HID] bf16
                  float* __restrict__ out_h, float* __restrict__ out_c) {
  extern __shared__ char smem[];
  u16* sh_h = (u16*)smem;                              // [BM][HID] 64KB
  u16* sh_x = (u16*)(smem + 65536);                    // [BM][IN ] 32KB
  u16* sh_s = (u16*)(smem + 65536 + 32768);            // [BM][HID] 64KB
  u16* sh_i = (u16*)(smem + 65536 + 32768 + 65536);    // [BM][HID] 64KB (stash)

  const int tid   = threadIdx.x;
  const int lane  = tid & 31;
  const int wave  = tid >> 5;
  const int mA    = (wave & 3) * 2;   // M tile pair base (rows mA*16 .. +32)
  const int nBase = (wave >> 2) * 64; // N quarter (4 tiles of 16)
  const int bBase = blockIdx.x * BM;

  // ---- phase 0: stage h_prev / x_t tile into LDS as bf16 (streaming loads) ----
  for (int i = tid * 4; i < BM * HID; i += THREADS * 4) {
    int r = i >> 8, c = i & (HID - 1);
    v4f v = __builtin_nontemporal_load(
        (const v4f*)(h_prev + (size_t)(bBase + r) * HID + c));
    v4u16 p = {f2bf_bits(v.x), f2bf_bits(v.y), f2bf_bits(v.z), f2bf_bits(v.w)};
    *(v4u16*)(sh_h + r * HID + c) = p;
  }
  for (int i = tid * 4; i < BM * IN; i += THREADS * 4) {
    int r = i >> 7, c = i & (IN - 1);
    v4f v = __builtin_nontemporal_load(
        (const v4f*)(x_t + (size_t)(bBase + r) * IN + c));
    v4u16 p = {f2bf_bits(v.x), f2bf_bits(v.y), f2bf_bits(v.z), f2bf_bits(v.w)};
    *(v4u16*)(sh_x + r * IN + c) = p;
  }
  __syncthreads();

  // C layout: VGPR r of lane L -> row rM0(mt)+r, col = nBase + nt*16 + (L&15)
  const int rOff = (lane >> 4) * 8;
  const int cLn  = lane & 15;

  // ---- phase 1: s_t = tanh(h@Wh0' + x@Wx0' + delta*Wst + b0) -> LDS bf16 ----
  v8f acc[MT][NT];
  zero_acc(acc);
  gemm_tile<HID>(acc, sh_h, Whb, lane, mA, nBase);
  gemm_tile<IN >(acc, sh_x, Wxb, lane, mA, nBase);

#pragma unroll
  for (int mt = 0; mt < MT; ++mt) {
    const int rM0 = (mA + mt) * 16 + rOff;
    float dlt[8];
#pragma unroll
    for (int r = 0; r < 8; ++r) dlt[r] = delta_t[bBase + rM0 + r];
#pragma unroll
    for (int nt = 0; nt < NT; ++nt) {
      int col = nBase + nt * 16 + cLn;
      float wst = Wst[col], b0 = bias[col];
#pragma unroll
      for (int r = 0; r < 8; ++r) {
        float sv = tanhf(acc[mt][nt][r] + dlt[r] * wst + b0);
        sh_s[(rM0 + r) * HID + col] = f2bf_bits(sv);
      }
    }
  }
  __syncthreads();

  // ---- phase 2: gates f(1), i(2), u(4), T(3), o(5); fold into c_t in regs ----
  v8f accC[MT][NT];

  // f: accC = sigmoid(f) * c_prev
  zero_acc(acc);
  gate_gemm(acc, 1, sh_h, sh_x, sh_s, Whb, Wxb, Wsb, lane, mA, nBase);
#pragma unroll
  for (int mt = 0; mt < MT; ++mt) {
    const int rM0 = (mA + mt) * 16 + rOff;
#pragma unroll
    for (int nt = 0; nt < NT; ++nt) {
      int col = nBase + nt * 16 + cLn;
      float bg = bias[1 * HID + col];
#pragma unroll
      for (int r = 0; r < 8; ++r) {
        size_t idx = (size_t)(bBase + rM0 + r) * HID + col;
        float cp = __builtin_nontemporal_load(c_prev + idx);
        accC[mt][nt][r] = sigm(acc[mt][nt][r] + bg) * cp;
      }
    }
  }

  // i: stash sigmoid(i) in LDS bf16 (same wave writes & reads -> no barrier)
  zero_acc(acc);
  gate_gemm(acc, 2, sh_h, sh_x, sh_s, Whb, Wxb, Wsb, lane, mA, nBase);
#pragma unroll
  for (int mt = 0; mt < MT; ++mt) {
    const int rM0 = (mA + mt) * 16 + rOff;
#pragma unroll
    for (int nt = 0; nt < NT; ++nt) {
      int col = nBase + nt * 16 + cLn;
      float bg = bias[2 * HID + col];
#pragma unroll
      for (int r = 0; r < 8; ++r)
        sh_i[(rM0 + r) * HID + col] = f2bf_bits(sigm(acc[mt][nt][r] + bg));
    }
  }

  // u (zeta): accC += stash * tanh(u)
  zero_acc(acc);
  gate_gemm(acc, 4, sh_h, sh_x, sh_s, Whb, Wxb, Wsb, lane, mA, nBase);
#pragma unroll
  for (int mt = 0; mt < MT; ++mt) {
    const int rM0 = (mA + mt) * 16 + rOff;
#pragma unroll
    for (int nt = 0; nt < NT; ++nt) {
      int col = nBase + nt * 16 + cLn;
      float bg = bias[4 * HID + col];
#pragma unroll
      for (int r = 0; r < 8; ++r) {
        float it = bf2f(sh_i[(rM0 + r) * HID + col]);
        accC[mt][nt][r] += it * tanhf(acc[mt][nt][r] + bg);
      }
    }
  }

  // T: accC += sigmoid(T) * s_t
  zero_acc(acc);
  gate_gemm(acc, 3, sh_h, sh_x, sh_s, Whb, Wxb, Wsb, lane, mA, nBase);
#pragma unroll
  for (int mt = 0; mt < MT; ++mt) {
    const int rM0 = (mA + mt) * 16 + rOff;
#pragma unroll
    for (int nt = 0; nt < NT; ++nt) {
      int col = nBase + nt * 16 + cLn;
      float bg = bias[3 * HID + col];
#pragma unroll
      for (int r = 0; r < 8; ++r) {
        float sv = bf2f(sh_s[(rM0 + r) * HID + col]);
        accC[mt][nt][r] += sigm(acc[mt][nt][r] + bg) * sv;
      }
    }
  }

  // o: h = sigmoid(o) * tanh(c_t); write h and c_t (streaming stores)
  zero_acc(acc);
  gate_gemm(acc, 5, sh_h, sh_x, sh_s, Whb, Wxb, Wsb, lane, mA, nBase);
#pragma unroll
  for (int mt = 0; mt < MT; ++mt) {
    const int rM0 = (mA + mt) * 16 + rOff;
#pragma unroll
    for (int nt = 0; nt < NT; ++nt) {
      int col = nBase + nt * 16 + cLn;
      float bg = bias[5 * HID + col];
#pragma unroll
      for (int r = 0; r < 8; ++r) {
        size_t idx = (size_t)(bBase + rM0 + r) * HID + col;
        float ct = accC[mt][nt][r];
        __builtin_nontemporal_store(ct, out_c + idx);
        __builtin_nontemporal_store(sigm(acc[mt][nt][r] + bg) * tanhf(ct),
                                    out_h + idx);
      }
    }
  }
}

// ---------- launch ----------
extern "C" void kernel_launch(void* const* d_in, const int* in_sizes, int n_in,
                              void* d_out, int out_size, void* d_ws, size_t ws_size,
                              hipStream_t stream) {
  const float* x_t    = (const float*)d_in[0];
  const float* delta  = (const float*)d_in[1];
  const float* h_prev = (const float*)d_in[2];
  const float* c_prev = (const float*)d_in[3];
  const float* Wh     = (const float*)d_in[4];
  const float* Wx     = (const float*)d_in[5];
  const float* Ws     = (const float*)d_in[6];
  const float* Wst    = (const float*)d_in[7];
  const float* bias   = (const float*)d_in[8];

  const int nWh = 6 * HID * HID;   // 393216
  const int nWx = 6 * HID * IN;    // 196608
  const int nWs = 5 * HID * HID;   // 327680
  u16* Whb = (u16*)d_ws;
  u16* Wxb = Whb + nWh;
  u16* Wsb = Wxb + nWx;

  cvt_f32_bf16<<<512, 256, 0, stream>>>(Wh, Whb, nWh);
  cvt_f32_bf16<<<256, 256, 0, stream>>>(Wx, Wxb, nWx);
  cvt_f32_bf16<<<512, 256, 0, stream>>>(Ws, Wsb, nWs);

  float* out_h = (float*)d_out;
  float* out_c = out_h + (size_t)BATCH * HID;

  size_t lds = 65536 + 32768 + 65536 + 65536;  // h + x + s + i-stash = 224 KB
  thlstm_fused<<<dim3(BATCH / BM), dim3(THREADS), lds, stream>>>(
      x_t, delta, h_prev, c_prev, Wst, bias, Whb, Wxb, Wsb, out_h, out_c);
}
